// SqueezeformerBlock_27212912787495
// MI455X (gfx1250) — matrix operand order repaired and checked by reading, not compiled, not run on hardware
//
#include <hip/hip_runtime.h>

// ---------------- types / helpers ----------------
typedef __bf16 bf16_t;
typedef __attribute__((ext_vector_type(16))) __bf16 v16bf;
typedef __attribute__((ext_vector_type(8)))  float  v8f;
typedef __attribute__((ext_vector_type(4)))  float  vf4;
typedef __attribute__((ext_vector_type(2)))  float  vf2;
typedef __attribute__((ext_vector_type(4)))  __bf16 v4bf;
typedef __attribute__((ext_vector_type(2)))  __bf16 v2bf;

#define S_LEN   2048
#define D_MODEL 256
#define NH      8
#define DHD     32
#define MROWS   4096   // B*S

// hardware f32->bf16 converts via vector convert (backend picks v_cvt_pk_bf16_f32)
__device__ __forceinline__ unsigned pack2(float lo, float hi) {
    vf2 f = { lo, hi };
    v2bf b = __builtin_convertvector(f, v2bf);
    return __builtin_bit_cast(unsigned, b);
}
__device__ __forceinline__ uint2 cvt4(float4 a) {
    vf4 f = { a.x, a.y, a.z, a.w };
    v4bf b = __builtin_convertvector(f, v4bf);
    return __builtin_bit_cast(uint2, b);
}
__device__ __forceinline__ uint4 pack8(float4 a, float4 b) {
    uint2 lo = cvt4(a), hi = cvt4(b);
    uint4 r; r.x = lo.x; r.y = lo.y; r.z = hi.x; r.w = hi.y;
    return r;
}

union frag16 { v16bf v; uint4 q[2]; unsigned u[8]; };

__device__ __forceinline__ v8f bwmma(v16bf a, v16bf b, v8f c) {
    return __builtin_amdgcn_wmma_f32_16x16x32_bf16(false, a, false, b, (short)0, c, false, false);
}

__device__ __forceinline__ float silu(float x) { return x / (1.f + __expf(-x)); }

// ---------------- generic WMMA GEMM: C = act(A@B + bias) + resid ----------------
// CTA tile 128x64, 256 threads = 8 waves (4x2), each wave 32x32 (4 WMMA accums:
// 2 A-frags x 2 B-frags, so B fragments are reused -> 4 wmma per 8 ds_load_b128).
// LDS holds packed bf16-pairs (uint): A row-major pair-packed; B pair-interleaved
// fragment-ready layout so every fragment is 2 x ds_load_b128.
__global__ void __launch_bounds__(256)
gemm_wmma(const float* __restrict__ A, const float* __restrict__ Bw,
          const float* __restrict__ bias, const float* __restrict__ resid,
          float* __restrict__ C, int M, int N, int Kd, int act)
{
    __shared__ unsigned Asu[128][20];   // [row][k/2] packed pairs (80B rows, 16B aligned)
    __shared__ unsigned Bpu[64][20];    // [n][lh*8+j] = pack(B[kb][n], B[kb+1][n]), kb=lh*16+2j

    const int tid  = threadIdx.x;
    const int wave = tid >> 5;
    const int lane = tid & 31;
    const int wm   = wave >> 1;          // 0..3 -> m offset wm*32
    const int wn   = wave & 1;           // 0..1 -> n offset wn*32
    const int lm   = lane & 15;
    const int lh   = lane >> 4;
    const int m0   = blockIdx.y * 128;
    const int n0   = blockIdx.x * 64;

    // staging coordinates
    const int ar = tid >> 1, ac = (tid & 1) * 16;         // A: row, col16 (16 floats/thread)
    const int br = tid >> 3, bc = (tid & 7) * 8;          // B: krow, col8
    const int arow = m0 + ar;
    const bool aval = (arow < M);
    const float* srcA = A + (size_t)arow * Kd + ac;
    const float* srcB = Bw + (size_t)br * N + n0 + bc;
    const int bj  = (br & 15) >> 1;                       // pair index within half
    const int blh = br >> 4;                              // K-half
    const int bhalf = br & 1;

    v8f acc00 = {}, acc01 = {}, acc10 = {}, acc11 = {};

    for (int kk = 0; kk < Kd; kk += 32) {
        {   // stage A 128x32 -> packed pairs, two uint4 stores per thread
            float4 a0 = {}, a1 = {}, a2 = {}, a3 = {};
            if (aval) {
                const float4* s4 = (const float4*)(srcA + kk);
                a0 = s4[0]; a1 = s4[1]; a2 = s4[2]; a3 = s4[3];
            }
            *(uint4*)&Asu[ar][(ac >> 1)]     = pack8(a0, a1);
            *(uint4*)&Asu[ar][(ac >> 1) + 4] = pack8(a2, a3);
        }
        {   // stage B 32x64 -> pair-interleaved fragment-ready layout
            const float4* s4 = (const float4*)(srcB + (size_t)kk * N);
            float4 b0 = s4[0], b1 = s4[1];
            v4bf c0 = __builtin_convertvector((vf4){b0.x, b0.y, b0.z, b0.w}, v4bf);
            v4bf c1 = __builtin_convertvector((vf4){b1.x, b1.y, b1.z, b1.w}, v4bf);
            bf16_t e[8] = { c0[0], c0[1], c0[2], c0[3], c1[0], c1[1], c1[2], c1[3] };
            #pragma unroll
            for (int i = 0; i < 8; ++i)
                ((bf16_t*)&Bpu[bc + i][blh * 8 + bj])[bhalf] = e[i];
        }
        if (kk + 32 < Kd) {   // prefetch next K-tile
            __builtin_prefetch((const void*)(srcA + kk + 32), 0, 0);
            __builtin_prefetch((const void*)(srcB + (size_t)(kk + 32) * N), 0, 0);
        }
        __syncthreads();

        frag16 a0f, a1f, b0f, b1f;
        a0f.q[0] = *(const uint4*)&Asu[wm*32 + lm][lh*8];            // rows wm*32..+15
        a0f.q[1] = *(const uint4*)&Asu[wm*32 + lm][lh*8 + 4];
        a1f.q[0] = *(const uint4*)&Asu[wm*32 + 16 + lm][lh*8];       // rows +16..+31
        a1f.q[1] = *(const uint4*)&Asu[wm*32 + 16 + lm][lh*8 + 4];
        b0f.q[0] = *(const uint4*)&Bpu[wn*32 + lm][lh*8];
        b0f.q[1] = *(const uint4*)&Bpu[wn*32 + lm][lh*8 + 4];
        b1f.q[0] = *(const uint4*)&Bpu[wn*32 + 16 + lm][lh*8];
        b1f.q[1] = *(const uint4*)&Bpu[wn*32 + 16 + lm][lh*8 + 4];

        acc00 = bwmma(a0f.v, b0f.v, acc00);
        acc01 = bwmma(a0f.v, b1f.v, acc01);
        acc10 = bwmma(a1f.v, b0f.v, acc10);
        acc11 = bwmma(a1f.v, b1f.v, acc11);
        __syncthreads();
    }

    #pragma unroll
    for (int ai = 0; ai < 2; ++ai) {
        #pragma unroll
        for (int g = 0; g < 8; ++g) {
            int row = m0 + wm*32 + ai*16 + g + 8*lh;
            if (row < M) {
                int col0 = n0 + wn*32 + lm;
                int col1 = col0 + 16;
                float v0 = ai ? acc10[g] : acc00[g];
                float v1 = ai ? acc11[g] : acc01[g];
                if (bias)     { v0 += bias[col0]; v1 += bias[col1]; }
                if (act == 1) { v0 = silu(v0);    v1 = silu(v1); }
                if (resid)    { v0 += resid[(size_t)row*N + col0];
                                v1 += resid[(size_t)row*N + col1]; }
                C[(size_t)row*N + col0] = v0;
                C[(size_t)row*N + col1] = v1;
            }
        }
    }
}

// ---------------- relative positional encoding ----------------
__global__ void pe_kernel(float* __restrict__ pe)
{
    int idx = blockIdx.x * 256 + threadIdx.x;
    const int total = (2*S_LEN - 1) * D_MODEL;
    if (idx >= total) return;
    int r = idx / D_MODEL, d = idx % D_MODEL;
    int i = d >> 1;
    float w   = __expf(-(float)(2*i) * (9.210340371976184f / (float)D_MODEL));
    float ang = (float)(S_LEN - 1 - r) * w;
    pe[idx] = (d & 1) ? __cosf(ang) : __sinf(ang);
}

// build an A/B-pattern fragment from 2x(2 float4) runs
__device__ __forceinline__ void frag_pack(frag16& f, float4 a0, float4 a1, float4 b0, float4 b1) {
    uint2 p0 = cvt4(a0), p1 = cvt4(a1), p2 = cvt4(b0), p3 = cvt4(b1);
    f.u[0] = p0.x; f.u[1] = p0.y; f.u[2] = p1.x; f.u[3] = p1.y;
    f.u[4] = p2.x; f.u[5] = p2.y; f.u[6] = p3.x; f.u[7] = p3.y;
}

// ---------------- fused rel-pos attention ----------------
// One block = one (b,h) x 16 query rows. Scores (16x2048) live in LDS.
__global__ void __launch_bounds__(256)
attn_kernel(const float* __restrict__ q, const float* __restrict__ kt,
            const float* __restrict__ vt, const float* __restrict__ p,
            const float* __restrict__ ub, const float* __restrict__ vb,
            float* __restrict__ ctx)
{
    extern __shared__ float smem[];
    float* sc     = smem;                      // [16][S_LEN]
    float* red    = smem + 16 * S_LEN;         // [8][16][32]
    float* rowsum = red  + 8 * 16 * 32;        // [16]

    const int tid  = threadIdx.x;
    const int wave = tid >> 5;
    const int lane = tid & 31;
    const int lm   = lane & 15;
    const int lh   = lane >> 4;
    const int nqt  = S_LEN / 16;
    const int bh   = blockIdx.x / nqt;
    const int i0   = (blockIdx.x % nqt) * 16;
    const int b    = bh / NH, h = bh % NH;
    const size_t base_bs = (size_t)b * S_LEN;
    const int hd   = h * DHD;
    const float scale = 0.17677669529663687f;  // 1/sqrt(32)

    // Q fragments: A-pattern runs are floats [lh*8 .. +7] and [16+lh*8 .. +7]
    frag16 aqc, aqp;
    {
        const float* qrow = q + (base_bs + i0 + lm) * D_MODEL + hd;
        float4 q0 = *(const float4*)(qrow + lh*8);
        float4 q1 = *(const float4*)(qrow + lh*8 + 4);
        float4 q2 = *(const float4*)(qrow + 16 + lh*8);
        float4 q3 = *(const float4*)(qrow + 16 + lh*8 + 4);
        float4 u0 = *(const float4*)(ub + hd + lh*8);
        float4 u1 = *(const float4*)(ub + hd + lh*8 + 4);
        float4 u2 = *(const float4*)(ub + hd + 16 + lh*8);
        float4 u3 = *(const float4*)(ub + hd + 16 + lh*8 + 4);
        float4 v0 = *(const float4*)(vb + hd + lh*8);
        float4 v1 = *(const float4*)(vb + hd + lh*8 + 4);
        float4 v2 = *(const float4*)(vb + hd + 16 + lh*8);
        float4 v3 = *(const float4*)(vb + hd + 16 + lh*8 + 4);
        frag_pack(aqc, make_float4(q0.x+u0.x,q0.y+u0.y,q0.z+u0.z,q0.w+u0.w),
                       make_float4(q1.x+u1.x,q1.y+u1.y,q1.z+u1.z,q1.w+u1.w),
                       make_float4(q2.x+u2.x,q2.y+u2.y,q2.z+u2.z,q2.w+u2.w),
                       make_float4(q3.x+u3.x,q3.y+u3.y,q3.z+u3.z,q3.w+u3.w));
        frag_pack(aqp, make_float4(q0.x+v0.x,q0.y+v0.y,q0.z+v0.z,q0.w+v0.w),
                       make_float4(q1.x+v1.x,q1.y+v1.y,q1.z+v1.z,q1.w+v1.w),
                       make_float4(q2.x+v2.x,q2.y+v2.y,q2.z+v2.z,q2.w+v2.w),
                       make_float4(q3.x+v3.x,q3.y+v3.y,q3.z+v3.z,q3.w+v3.w));
    }

    // ---- Phase 1: scores = (content + shifted pos) * scale, into LDS ----
    for (int jt = wave; jt < nqt; jt += 8) {
        int j0 = jt * 16;
        int rbase = S_LEN - 16 - i0 + j0;      // r=(S-1)-s+t  =>  band col = 15 - m + n
        frag16 bk, bp0, bp1;
        {
            const float* kr = kt + (base_bs + j0 + lm) * D_MODEL + hd;
            frag_pack(bk, *(const float4*)(kr + lh*8),      *(const float4*)(kr + lh*8 + 4),
                          *(const float4*)(kr + 16 + lh*8), *(const float4*)(kr + 16 + lh*8 + 4));
            const float* p0 = p + (size_t)(rbase + lm) * D_MODEL + hd;
            frag_pack(bp0, *(const float4*)(p0 + lh*8),      *(const float4*)(p0 + lh*8 + 4),
                           *(const float4*)(p0 + 16 + lh*8), *(const float4*)(p0 + 16 + lh*8 + 4));
            const float* p1 = p + (size_t)(rbase + 16 + lm) * D_MODEL + hd;
            frag_pack(bp1, *(const float4*)(p1 + lh*8),      *(const float4*)(p1 + lh*8 + 4),
                           *(const float4*)(p1 + 16 + lh*8), *(const float4*)(p1 + 16 + lh*8 + 4));
        }
        v8f cc = {}, t0 = {}, t1 = {};
        cc = bwmma(aqc.v, bk.v,  cc);
        t0 = bwmma(aqp.v, bp0.v, t0);
        t1 = bwmma(aqp.v, bp1.v, t1);
        #pragma unroll
        for (int g = 0; g < 8; ++g) {
            int ml = g + 8 * lh;               // query row within tile
            int c  = 15 - ml + lm;             // band column 0..30
            int sl = (c & 15) + 16 * lh;       // source lane in same half
            float pv0 = __shfl(t0[g], sl, 32);
            float pv1 = __shfl(t1[g], sl, 32);
            float pv  = (c < 16) ? pv0 : pv1;
            sc[ml * S_LEN + j0 + lm] = (cc[g] + pv) * scale;
        }
    }
    __syncthreads();

    // ---- Phase 2: softmax numerators in LDS (float4-wide), row sums aside ----
    for (int r = wave * 2; r < wave * 2 + 2; ++r) {
        float* row = sc + r * S_LEN;
        float mx = -3.4e38f;
        for (int c = lane * 4; c < S_LEN; c += 128) {
            float4 vv = *(const float4*)(row + c);
            mx = fmaxf(mx, fmaxf(fmaxf(vv.x, vv.y), fmaxf(vv.z, vv.w)));
        }
        #pragma unroll
        for (int off = 16; off > 0; off >>= 1) mx = fmaxf(mx, __shfl_xor(mx, off, 32));
        float sum = 0.f;
        for (int c = lane * 4; c < S_LEN; c += 128) {
            float4 vv = *(const float4*)(row + c);
            vv.x = __expf(vv.x - mx); vv.y = __expf(vv.y - mx);
            vv.z = __expf(vv.z - mx); vv.w = __expf(vv.w - mx);
            *(float4*)(row + c) = vv;
            sum += vv.x + vv.y + vv.z + vv.w;
        }
        #pragma unroll
        for (int off = 16; off > 0; off >>= 1) sum += __shfl_xor(sum, off, 32);
        if (lane == 0) rowsum[r] = sum;
    }
    __syncthreads();

    // ---- Phase 3: O = P @ V (waves split K=2048, partials reduced in LDS) ----
    v8f o0 = {}, o1 = {};
    for (int kc = wave; kc < S_LEN / 32; kc += 8) {
        int t0k = kc * 32;
        frag16 ap, bv0, bv1;
        {
            const float* sr = sc + lm * S_LEN + t0k;
            frag_pack(ap, *(const float4*)(sr + lh*8),      *(const float4*)(sr + lh*8 + 4),
                          *(const float4*)(sr + 16 + lh*8), *(const float4*)(sr + 16 + lh*8 + 4));
        }
        #pragma unroll
        for (int j = 0; j < 8; ++j) {
            int kb = lh * 8 + 2 * j + (j & 4) * 2;
            const float* vr0 = vt + (base_bs + t0k + kb) * D_MODEL + hd;
            const float* vr1 = vr0 + D_MODEL;
            bv0.u[j] = pack2(vr0[lm],      vr1[lm]);
            bv1.u[j] = pack2(vr0[16 + lm], vr1[16 + lm]);
        }
        o0 = bwmma(ap.v, bv0.v, o0);
        o1 = bwmma(ap.v, bv1.v, o1);
    }
    #pragma unroll
    for (int g = 0; g < 8; ++g) {
        int ml = g + 8 * lh;
        red[(wave * 16 + ml) * 32 + lm]      = o0[g];
        red[(wave * 16 + ml) * 32 + 16 + lm] = o1[g];
    }
    __syncthreads();
    for (int idx = tid; idx < 16 * 32; idx += 256) {
        int ml = idx >> 5, n = idx & 31;
        float s = 0.f;
        #pragma unroll
        for (int w = 0; w < 8; ++w) s += red[(w * 16 + ml) * 32 + n];
        s /= rowsum[ml];
        ctx[(base_bs + i0 + ml) * D_MODEL + hd + n] = s;
    }
}

// ---------------- LayerNorm over D=256 (one block per row) ----------------
__global__ void __launch_bounds__(256)
ln_kernel(const float* __restrict__ in, const float* __restrict__ g,
          const float* __restrict__ bta, float* __restrict__ out)
{
    __shared__ float rbuf[8];
    const int row = blockIdx.x, tid = threadIdx.x;
    const int wave = tid >> 5, lane = tid & 31;
    float v = in[(size_t)row * D_MODEL + tid];

    float s = v;
    #pragma unroll
    for (int off = 16; off > 0; off >>= 1) s += __shfl_xor(s, off, 32);
    if (lane == 0) rbuf[wave] = s;
    __syncthreads();
    float t = (lane < 8) ? rbuf[lane] : 0.f;
    #pragma unroll
    for (int off = 4; off > 0; off >>= 1) t += __shfl_xor(t, off, 32);
    float mean = __shfl(t, 0, 32) * (1.f / 256.f);

    float d = v - mean;
    float s2 = d * d;
    #pragma unroll
    for (int off = 16; off > 0; off >>= 1) s2 += __shfl_xor(s2, off, 32);
    __syncthreads();
    if (lane == 0) rbuf[wave] = s2;
    __syncthreads();
    float t2 = (lane < 8) ? rbuf[lane] : 0.f;
    #pragma unroll
    for (int off = 4; off > 0; off >>= 1) t2 += __shfl_xor(t2, off, 32);
    float var = __shfl(t2, 0, 32) * (1.f / 256.f);

    out[(size_t)row * D_MODEL + tid] = d * rsqrtf(var + 1e-5f) * g[tid] + bta[tid];
}

// ---------------- GLU: out = a * sigmoid(g), halves of [M,2D] ----------------
__global__ void glu_kernel(const float* __restrict__ in, float* __restrict__ out)
{
    int idx = blockIdx.x * 256 + threadIdx.x;   // M*D total
    int r = idx >> 8, c = idx & 255;
    float a = in[(size_t)r * 512 + c];
    float gg = in[(size_t)r * 512 + 256 + c];
    out[idx] = a / (1.f + __expf(-gg));
}

// ---------------- depthwise conv K=5 (SAME) + BN affine + SiLU ----------------
__global__ void dwconv_kernel(const float* __restrict__ in, const float* __restrict__ w,
                              const float* __restrict__ g, const float* __restrict__ bta,
                              float* __restrict__ out)
{
    int idx = blockIdx.x * 256 + threadIdx.x;   // B*S*D
    int d = idx & 255;
    int s = (idx >> 8) & (S_LEN - 1);
    int b = idx >> 19;
    float acc = 0.f;
    #pragma unroll
    for (int kk = 0; kk < 5; ++kk) {
        int ss = s + kk - 2;
        if (ss >= 0 && ss < S_LEN)
            acc += in[((size_t)(b * S_LEN + ss) << 8) + d] * w[d * 5 + kk];
    }
    const float bnscale = 0.9999950000374997f;  // rsqrt(1 + 1e-5)
    float hh = acc * bnscale * g[d] + bta[d];
    out[idx] = silu(hh);
}

// ---------------- launch ----------------
extern "C" void kernel_launch(void* const* d_in, const int* in_sizes, int n_in,
                              void* d_out, int out_size, void* d_ws, size_t ws_size,
                              hipStream_t stream)
{
    const float* x      = (const float*)d_in[0];
    const float* Wq     = (const float*)d_in[1];
    const float* bq     = (const float*)d_in[2];
    const float* Wk     = (const float*)d_in[3];
    const float* bk     = (const float*)d_in[4];
    const float* Wv     = (const float*)d_in[5];
    const float* bv     = (const float*)d_in[6];
    const float* Wpos   = (const float*)d_in[7];
    const float* u_bias = (const float*)d_in[8];
    const float* v_bias = (const float*)d_in[9];
    const float* Wo     = (const float*)d_in[10];
    const float* bo     = (const float*)d_in[11];
    const float* ln1_g  = (const float*)d_in[12];
    const float* ln1_b  = (const float*)d_in[13];
    const float* ff1_W1 = (const float*)d_in[14];
    const float* ff1_b1 = (const float*)d_in[15];
    const float* ff1_W2 = (const float*)d_in[16];
    const float* ff1_b2 = (const float*)d_in[17];
    const float* ln2_g  = (const float*)d_in[18];
    const float* ln2_b  = (const float*)d_in[19];
    const float* pw1_W  = (const float*)d_in[20];
    const float* pw1_b  = (const float*)d_in[21];
    const float* dw_W   = (const float*)d_in[22];
    const float* bn_g   = (const float*)d_in[23];
    const float* bn_b   = (const float*)d_in[24];
    const float* pw2_W  = (const float*)d_in[25];
    const float* pw2_b  = (const float*)d_in[26];
    const float* ln3_g  = (const float*)d_in[27];
    const float* ln3_b  = (const float*)d_in[28];
    const float* ff2_W1 = (const float*)d_in[29];
    const float* ff2_b1 = (const float*)d_in[30];
    const float* ff2_W2 = (const float*)d_in[31];
    const float* ff2_b2 = (const float*)d_in[32];
    const float* ln4_g  = (const float*)d_in[33];
    const float* ln4_b  = (const float*)d_in[34];

    float* ws = (float*)d_ws;
    const size_t M1 = 1024 * 1024;       // 1M floats
    float* WS_H  = ws;                   // 4M : FFN mid [4096,1024]
    float* WS_H2 = ws + 4  * M1;         // 2M : pw1 out [4096,512]
    float* WS_PE = ws + 6  * M1;         // 1M : pe, later reused as ctx
    float* WS_Q  = ws + 7  * M1;         // 1M : q, later reused as conv out
    float* WS_K  = ws + 8  * M1;         // 1M : k, later reused as GLU out
    float* WS_V  = ws + 9  * M1;         // 1M : v
    float* WS_P  = ws + 10 * M1;         // 1M : pos projection [4095,256] (+pad row)
    float* WS_T  = ws + 11 * M1;         // 1M : pre-LN sums
    float* WS_X1 = ws + 12 * M1;
    float* WS_X2 = ws + 13 * M1;
    float* WS_X3 = ws + 14 * M1;

    dim3 blk(256);
    auto gemm = [&](const float* A, const float* Bw, const float* bias,
                    const float* resid, float* C, int M, int N, int K, int act) {
        dim3 grid(N / 64, (M + 127) / 128);
        gemm_wmma<<<grid, blk, 0, stream>>>(A, Bw, bias, resid, C, M, N, K, act);
    };

    // positional encoding + projection
    int peTot = (2 * S_LEN - 1) * D_MODEL;
    pe_kernel<<<(peTot + 255) / 256, blk, 0, stream>>>(WS_PE);
    gemm(WS_PE, Wpos, nullptr, nullptr, WS_P, 2 * S_LEN - 1, D_MODEL, D_MODEL, 0);

    // QKV projections
    gemm(x, Wq, bq, nullptr, WS_Q, MROWS, D_MODEL, D_MODEL, 0);
    gemm(x, Wk, bk, nullptr, WS_K, MROWS, D_MODEL, D_MODEL, 0);
    gemm(x, Wv, bv, nullptr, WS_V, MROWS, D_MODEL, D_MODEL, 0);

    // fused rel-pos attention (scores never leave LDS: 144KB dyn-LDS of 320KB/WGP)
    size_t shmem = (size_t)(16 * S_LEN + 8 * 16 * 32 + 16) * sizeof(float);
    attn_kernel<<<dim3(2 * NH * (S_LEN / 16)), blk, shmem, stream>>>(
        WS_Q, WS_K, WS_V, WS_P, u_bias, v_bias, WS_PE /* ctx */);

    // Wo + residual, LN1
    gemm(WS_PE, Wo, bo, x, WS_T, MROWS, D_MODEL, D_MODEL, 0);
    ln_kernel<<<MROWS, blk, 0, stream>>>(WS_T, ln1_g, ln1_b, WS_X1);

    // FFN1 (SiLU fused) + residual, LN2
    gemm(WS_X1, ff1_W1, ff1_b1, nullptr, WS_H, MROWS, 1024, D_MODEL, 1);
    gemm(WS_H, ff1_W2, ff1_b2, WS_X1, WS_T, MROWS, D_MODEL, 1024, 0);
    ln_kernel<<<MROWS, blk, 0, stream>>>(WS_T, ln2_g, ln2_b, WS_X2);

    // Conv module: pw1 -> GLU -> dwconv+BN+SiLU -> pw2 + residual, LN3
    gemm(WS_X2, pw1_W, pw1_b, nullptr, WS_H2, MROWS, 512, D_MODEL, 0);
    glu_kernel<<<MROWS * D_MODEL / 256, blk, 0, stream>>>(WS_H2, WS_K);
    dwconv_kernel<<<MROWS * D_MODEL / 256, blk, 0, stream>>>(WS_K, dw_W, bn_g, bn_b, WS_Q);
    gemm(WS_Q, pw2_W, pw2_b, WS_X2, WS_T, MROWS, D_MODEL, D_MODEL, 0);
    ln_kernel<<<MROWS, blk, 0, stream>>>(WS_T, ln3_g, ln3_b, WS_X3);

    // FFN2 + residual, LN4 -> output
    gemm(WS_X3, ff2_W1, ff2_b1, nullptr, WS_H, MROWS, 1024, D_MODEL, 1);
    gemm(WS_H, ff2_W2, ff2_b2, WS_X3, WS_T, MROWS, D_MODEL, 1024, 0);
    ln_kernel<<<MROWS, blk, 0, stream>>>(WS_T, ln4_g, ln4_b, (float*)d_out);
}
